// RNABlock_32469952758245
// MI455X (gfx1250) — matrix operand order repaired and checked by reading, not compiled, not run on hardware
//
#include <hip/hip_runtime.h>
#include <hip/hip_bf16.h>
#include <math.h>
#include <stdint.h>

typedef float    v2f  __attribute__((ext_vector_type(2)));
typedef float    v8f  __attribute__((ext_vector_type(8)));
typedef _Float16 v16h __attribute__((ext_vector_type(16)));

#define NEG_INF (-1e30f)

static inline int ceildiv_ll(long long a, long long b) { return (int)((a + b - 1) / b); }

// Low 32 bits of a flat pointer into LDS == LDS byte address (ISA 10.2 aperture),
// which is what GLOBAL_LOAD_ASYNC_TO_LDS's VDST wants.
__device__ __forceinline__ unsigned lds_addr32(const void* p)
{ return (unsigned)(uintptr_t)p; }

__device__ __forceinline__ void async_load_b32(unsigned lds_dst, const void* gsrc)
{
    unsigned long long ga = (unsigned long long)(uintptr_t)gsrc;
    asm volatile("global_load_async_to_lds_b32 %0, %1, off"
                 :: "v"(lds_dst), "v"(ga) : "memory");
}
__device__ __forceinline__ void wait_asynccnt0()
{
    asm volatile("s_wait_asynccnt 0x0" ::: "memory");
}

// ======================================================================
// WMMA GEMM, FP32 path: D[m,n] = scale * sum_k A[m,k]*B[k,n] + bias[m]
// Strided A/B/D so transposed views are free. One wave owns a 16x16 D
// tile; 8 waves/block share a 64x16 B panel staged in LDS — via the
// gfx1250 async global->LDS engine (ASYNCcnt) when fully in-bounds,
// or a branchless clamp+select loop otherwise. Inner loop is pure
// global A-load + LDS B-read + V_WMMA_F32_16X16X4_F32 in the full-tile fast path.
// ======================================================================
__global__ __launch_bounds__(256) void k_gemm_f32(
    const float* __restrict__ A, long long a_bs, int a_rs, int a_cs,
    const float* __restrict__ Bm, long long b_bs, int b_rs, int b_cs,
    float* __restrict__ D, long long d_bs, int d_rs, int d_cs,
    const float* __restrict__ bias, int M, int N, int K,
    float scale, int relu)
{
    __shared__ float lb[64 * 16];
    const int wave = threadIdx.x >> 5;
    const int lane = threadIdx.x & 31;
    const int lm = lane & 15, lh = lane >> 4;
    const int m0 = (blockIdx.y * 8 + wave) * 16;
    const int n0 = blockIdx.x * 16;
    const int bz = blockIdx.z;
    const float* Ab = A + (long long)bz * a_bs;
    const float* Bb = Bm + (long long)bz * b_bs;
    const int mA = m0 + lm;
    const int mC = (mA < M) ? mA : (M - 1);   // clamped row for branchless loads
    v8f acc;
#pragma unroll
    for (int i = 0; i < 8; ++i) acc[i] = 0.0f;

    for (int k0 = 0; k0 < K; k0 += 64) {
        const bool bfull = (k0 + 64 <= K) && (n0 + 16 <= N);
        if (bfull) {
            // async global->LDS staging of the 64x16 B panel (4 dwords/thread)
            for (int i = threadIdx.x; i < 64 * 16; i += 256) {
                const float* src = Bb + (long long)(k0 + (i >> 4)) * b_rs
                                      + (long long)(n0 + (i & 15)) * b_cs;
                async_load_b32(lds_addr32(&lb[i]), src);
            }
            wait_asynccnt0();
        } else {
            for (int i = threadIdx.x; i < 64 * 16; i += 256) {
                int kk = k0 + (i >> 4), nn = n0 + (i & 15);
                int kc = (kk < K) ? kk : (K - 1);
                int nc = (nn < N) ? nn : (N - 1);
                float v = Bb[(long long)kc * b_rs + (long long)nc * b_cs];
                lb[i] = (kk < K && nn < N) ? v : 0.0f;
            }
        }
        __syncthreads();
        if (k0 + 64 < K)
            __builtin_prefetch(&Bb[(long long)(k0 + 64) * b_rs + (long long)n0 * b_cs], 0, 1);

        const bool afull = (m0 + 16 <= M) && (k0 + 64 <= K);
        if (afull) {
#pragma unroll
            for (int kk = 0; kk < 64; kk += 4) {
                int ka = k0 + kk + lh * 2;
                v2f a, b;
                a.x = Ab[(long long)mA * a_rs + (long long)(ka    ) * a_cs];
                a.y = Ab[(long long)mA * a_rs + (long long)(ka + 1) * a_cs];
                b.x = lb[(kk + lh * 2    ) * 16 + lm];
                b.y = lb[(kk + lh * 2 + 1) * 16 + lm];
                acc = __builtin_amdgcn_wmma_f32_16x16x4_f32(false, a, false, b, (short)0, acc, false, false);
            }
        } else {
#pragma unroll
            for (int kk = 0; kk < 64; kk += 4) {
                int ka = k0 + kk + lh * 2;
                int k1 = (ka     < K) ? ka     : (K - 1);
                int k2 = (ka + 1 < K) ? ka + 1 : (K - 1);
                float v0 = Ab[(long long)mC * a_rs + (long long)k1 * a_cs];
                float v1 = Ab[(long long)mC * a_rs + (long long)k2 * a_cs];
                v2f a, b;
                a.x = (mA < M && ka     < K) ? v0 : 0.0f;
                a.y = (mA < M && ka + 1 < K) ? v1 : 0.0f;
                b.x = lb[(kk + lh * 2    ) * 16 + lm];
                b.y = lb[(kk + lh * 2 + 1) * 16 + lm];
                acc = __builtin_amdgcn_wmma_f32_16x16x4_f32(false, a, false, b, (short)0, acc, false, false);
            }
        }
        __syncthreads();
    }
    const int n = n0 + lm;
    if (n < N) {
        float* Db = D + (long long)bz * d_bs;
#pragma unroll
        for (int v = 0; v < 8; ++v) {
            int m = m0 + v + lh * 8;
            if (m < M) {
                float r = acc[v] * scale + (bias ? bias[m] : 0.0f);
                if (relu) r = fmaxf(r, 0.0f);
                Db[(long long)m * d_rs + (long long)n * d_cs] = r;
            }
        }
    }
}

// ======================================================================
// WMMA GEMM, FP16 inputs / FP32 accumulate (V_WMMA_F32_16X16X32_F16).
// Attention score (K=32 -> 1 WMMA/tile) and score x V. Async B-panel
// staging engages when b_cs==1 (pairs of halves = one b32 async load).
// ======================================================================
__global__ __launch_bounds__(256) void k_gemm_f16(
    const _Float16* __restrict__ A, long long a_bs, int a_rs, int a_cs,
    const _Float16* __restrict__ Bm, long long b_bs, int b_rs, int b_cs,
    float* __restrict__ D, long long d_bs, int d_rs, int d_cs,
    const float* __restrict__ bias, int M, int N, int K,
    float scale, int relu)
{
    __shared__ _Float16 lb[64 * 16];
    const int wave = threadIdx.x >> 5;
    const int lane = threadIdx.x & 31;
    const int lm = lane & 15, lh = lane >> 4;
    const int m0 = (blockIdx.y * 8 + wave) * 16;
    const int n0 = blockIdx.x * 16;
    const int bz = blockIdx.z;
    const _Float16* Ab = A + (long long)bz * a_bs;
    const _Float16* Bb = Bm + (long long)bz * b_bs;
    const int mA = m0 + lm;
    const int mC = (mA < M) ? mA : (M - 1);
    v8f acc;
#pragma unroll
    for (int i = 0; i < 8; ++i) acc[i] = 0.0f;

    for (int k0 = 0; k0 < K; k0 += 64) {
        const bool bfull = (k0 + 64 <= K) && (n0 + 16 <= N) && (b_cs == 1);
        if (bfull) {
            // 512 dword async loads: each moves 2 consecutive halves of a panel row
            for (int i = threadIdx.x; i < 512; i += 256) {
                int kl = i >> 3;            // local k row (0..63)
                int np = (i & 7) * 2;       // even n within row
                const _Float16* src = Bb + (long long)(k0 + kl) * b_rs + (n0 + np);
                async_load_b32(lds_addr32(&lb[kl * 16 + np]), src);
            }
            wait_asynccnt0();
        } else {
            for (int i = threadIdx.x; i < 64 * 16; i += 256) {
                int kk = k0 + (i >> 4), nn = n0 + (i & 15);
                int kc = (kk < K) ? kk : (K - 1);
                int nc = (nn < N) ? nn : (N - 1);
                _Float16 v = Bb[(long long)kc * b_rs + (long long)nc * b_cs];
                lb[i] = (kk < K && nn < N) ? v : (_Float16)0.0f;
            }
        }
        __syncthreads();
        if (k0 + 64 < K)
            __builtin_prefetch(&Bb[(long long)(k0 + 64) * b_rs + (long long)n0 * b_cs], 0, 1);

        const bool afull = (m0 + 16 <= M) && (k0 + 64 <= K);
        if (afull) {
#pragma unroll
            for (int kk = 0; kk < 64; kk += 32) {
                v16h a, b;
#pragma unroll
                for (int j = 0; j < 8; ++j) {
                    int kb = k0 + kk + (j & 3) * 2 + (j >> 2) * 16 + lh * 8;
                    a[2 * j]     = Ab[(long long)mA * a_rs + (long long)(kb    ) * a_cs];
                    a[2 * j + 1] = Ab[(long long)mA * a_rs + (long long)(kb + 1) * a_cs];
                }
#pragma unroll
                for (int j = 0; j < 8; ++j) {
                    int kb = kk + (j & 3) * 2 + (j >> 2) * 16 + lh * 8;
                    b[2 * j]     = lb[(kb    ) * 16 + lm];
                    b[2 * j + 1] = lb[(kb + 1) * 16 + lm];
                }
                acc = __builtin_amdgcn_wmma_f32_16x16x32_f16(false, a, false, b, (short)0, acc, false, false);
            }
        } else {
#pragma unroll
            for (int kk = 0; kk < 64; kk += 32) {
                v16h a, b;
#pragma unroll
                for (int j = 0; j < 8; ++j) {
                    int kb = k0 + kk + (j & 3) * 2 + (j >> 2) * 16 + lh * 8;
                    int c0 = (kb     < K) ? kb     : (K - 1);
                    int c1 = (kb + 1 < K) ? kb + 1 : (K - 1);
                    _Float16 v0 = Ab[(long long)mC * a_rs + (long long)c0 * a_cs];
                    _Float16 v1 = Ab[(long long)mC * a_rs + (long long)c1 * a_cs];
                    a[2 * j]     = (mA < M && kb     < K) ? v0 : (_Float16)0.0f;
                    a[2 * j + 1] = (mA < M && kb + 1 < K) ? v1 : (_Float16)0.0f;
                }
#pragma unroll
                for (int j = 0; j < 8; ++j) {
                    int kb = kk + (j & 3) * 2 + (j >> 2) * 16 + lh * 8;
                    b[2 * j]     = lb[(kb    ) * 16 + lm];
                    b[2 * j + 1] = lb[(kb + 1) * 16 + lm];
                }
                acc = __builtin_amdgcn_wmma_f32_16x16x32_f16(false, a, false, b, (short)0, acc, false, false);
            }
        }
        __syncthreads();
    }
    const int n = n0 + lm;
    if (n < N) {
        float* Db = D + (long long)bz * d_bs;
#pragma unroll
        for (int v = 0; v < 8; ++v) {
            int m = m0 + v + lh * 8;
            if (m < M) {
                float r = acc[v] * scale + (bias ? bias[m] : 0.0f);
                if (relu) r = fmaxf(r, 0.0f);
                Db[(long long)m * d_rs + (long long)n * d_cs] = r;
            }
        }
    }
}

// ======================================================================
// Stride-generic per-channel statistics (mean, biased var).
// perBatch=1 -> instance norm stats per (b, ch); else batch-norm stats per ch.
// ======================================================================
__global__ __launch_bounds__(256) void k_stats(
    const float* __restrict__ x, long long sb, long long sc, long long sl,
    int Bn, int L, int perBatch, float* __restrict__ mean, float* __restrict__ var, int CH)
{
    __shared__ float s1[256];
    __shared__ float s2[256];
    int ch = blockIdx.x;
    int bi = blockIdx.y;
    float a = 0.0f, b2 = 0.0f;
    if (perBatch) {
        const float* base = x + (long long)bi * sb + (long long)ch * sc;
        for (int l = threadIdx.x; l < L; l += 256) {
            float v = base[(long long)l * sl];
            a += v; b2 += v * v;
        }
    } else {
        for (int bb = 0; bb < Bn; ++bb) {
            const float* base = x + (long long)bb * sb + (long long)ch * sc;
            for (int l = threadIdx.x; l < L; l += 256) {
                float v = base[(long long)l * sl];
                a += v; b2 += v * v;
            }
        }
    }
    s1[threadIdx.x] = a; s2[threadIdx.x] = b2;
    __syncthreads();
    for (int o = 128; o > 0; o >>= 1) {
        if (threadIdx.x < o) { s1[threadIdx.x] += s1[threadIdx.x + o]; s2[threadIdx.x] += s2[threadIdx.x + o]; }
        __syncthreads();
    }
    if (threadIdx.x == 0) {
        float cnt = perBatch ? (float)L : (float)L * (float)Bn;
        float m = s1[0] / cnt;
        int idx = perBatch ? bi * CH + ch : ch;
        mean[idx] = m;
        var[idx] = s2[0] / cnt - m * m;
    }
}

__global__ void k_norm_apply(
    const float* __restrict__ x, float* __restrict__ y,
    long long sb, long long sc, long long sl, int CH, int L,
    const float* __restrict__ mean, const float* __restrict__ var,
    int perBatch, float eps, const float* __restrict__ g, const float* __restrict__ bt, int relu)
{
    int l = blockIdx.x * blockDim.x + threadIdx.x;
    int ch = blockIdx.y;
    int bi = blockIdx.z;
    if (l >= L) return;
    int si = perBatch ? bi * CH + ch : ch;
    long long off = (long long)bi * sb + (long long)ch * sc + (long long)l * sl;
    float val = (x[off] - mean[si]) * rsqrtf(var[si] + eps);
    if (g) val = val * g[ch] + bt[ch];
    if (relu) val = fmaxf(val, 0.0f);
    y[off] = val;
}

// ======================================================================
// Attention helpers
// ======================================================================
__global__ void k_rowsumsq(const float* __restrict__ x, float* __restrict__ xx, int C, int N, long long n)
{
    long long i = blockIdx.x * 256ll + threadIdx.x;
    if (i >= n) return;
    int b = (int)(i / N), col = (int)(i % N);
    const float* p = x + (long long)b * C * N + col;
    float s = 0.0f;
    for (int c = 0; c < C; ++c) { float v = p[(long long)c * N]; s += v * v; }
    xx[i] = s;
}

__global__ void k_pd(float* __restrict__ G, const float* __restrict__ xx, int N, long long n)
{   // in place: pd = 2*gram - xx[n] - xx[m] (gram already scaled by 2 in GEMM)
    long long i = blockIdx.x * 256ll + threadIdx.x;
    if (i >= n) return;
    long long NN = (long long)N * N;
    int b = (int)(i / NN);
    long long r = i % NN;
    int rr = (int)(r / N), cc = (int)(r % N);
    G[i] = G[i] - xx[b * N + rr] - xx[b * N + cc];
}

// One wave per row: iterative argmax selection of knn entries (wave32 ballot).
#define TKC 63
__global__ __launch_bounds__(256) void k_topk_mask(
    const float* __restrict__ pd, float* __restrict__ mask, int N, int knn, int rows)
{
    int wave = threadIdx.x >> 5, lane = threadIdx.x & 31;
    int row = blockIdx.x * 8 + wave;
    if (row >= rows) return;
    const float* p = pd + (long long)row * N;
    float vals[TKC];
#pragma unroll
    for (int i = 0; i < TKC; ++i) {
        int idx = lane + i * 32;
        vals[i] = (idx < N) ? p[idx] : NEG_INF;
    }
    unsigned long long sel = 0ull;
    for (int t = 0; t < knn; ++t) {
        float lmx = NEG_INF; int li = 0;
#pragma unroll
        for (int i = 0; i < TKC; ++i)
            if (vals[i] > lmx) { lmx = vals[i]; li = i; }
        float m = lmx;
        for (int o = 16; o > 0; o >>= 1) m = fmaxf(m, __shfl_xor(m, o, 32));
        unsigned long long ball = __ballot(lmx == m);
        int owner = __ffsll(ball) - 1;
        if (lane == owner) { sel |= (1ull << li); vals[li] = NEG_INF; }
    }
    float* mrow = mask + (long long)row * N;
#pragma unroll
    for (int i = 0; i < TKC; ++i) {
        int idx = lane + i * 32;
        if (idx < N) mrow[idx] = ((sel >> i) & 1ull) ? 1.0f : 0.0f;
    }
}

__global__ void k_masksym(const float* __restrict__ m, float* __restrict__ o, int N, long long n)
{
    long long i = blockIdx.x * 256ll + threadIdx.x;
    if (i >= n) return;
    long long NN = (long long)N * N;
    int b = (int)(i / NN);
    long long r = i % NN;
    int rr = (int)(r / N), cc = (int)(r % N);
    o[i] = m[i] * m[(long long)b * NN + (long long)cc * N + rr];
}

__global__ __launch_bounds__(256) void k_mask_softmax(
    float* __restrict__ sc, const float* __restrict__ mask, int N)
{
    __shared__ float sh[256];
    long long row = blockIdx.x;
    float* p = sc + row * N;
    const float* mk = mask + row * N;
    float mx = NEG_INF;
    for (int j = threadIdx.x; j < N; j += 256) {
        float m = mk[j];
        float v = p[j] * m - 1e9f * (1.0f - m);
        mx = fmaxf(mx, v);
    }
    sh[threadIdx.x] = mx; __syncthreads();
    for (int o = 128; o > 0; o >>= 1) { if (threadIdx.x < o) sh[threadIdx.x] = fmaxf(sh[threadIdx.x], sh[threadIdx.x + o]); __syncthreads(); }
    mx = sh[0]; __syncthreads();
    float s = 0.0f;
    for (int j = threadIdx.x; j < N; j += 256) {
        float m = mk[j];
        float v = p[j] * m - 1e9f * (1.0f - m);
        float e = __expf(v - mx);
        p[j] = e; s += e;
    }
    sh[threadIdx.x] = s; __syncthreads();
    for (int o = 128; o > 0; o >>= 1) { if (threadIdx.x < o) sh[threadIdx.x] += sh[threadIdx.x + o]; __syncthreads(); }
    float inv = 1.0f / sh[0]; __syncthreads();
    for (int j = threadIdx.x; j < N; j += 256) p[j] *= inv;
}

// generic strided softmax: row r base = (r/rinner)*s_outer + (r%rinner)*s_inner
__global__ __launch_bounds__(256) void k_softmax_stride(
    float* __restrict__ x, int R, int rinner, long long s_outer, long long s_inner,
    int L, long long estride)
{
    __shared__ float sh[256];
    int r = blockIdx.x;
    if (r >= R) return;
    float* p = x + (long long)(r / rinner) * s_outer + (long long)(r % rinner) * s_inner;
    float mx = NEG_INF;
    for (int j = threadIdx.x; j < L; j += 256) mx = fmaxf(mx, p[j * estride]);
    sh[threadIdx.x] = mx; __syncthreads();
    for (int o = 128; o > 0; o >>= 1) { if (threadIdx.x < o) sh[threadIdx.x] = fmaxf(sh[threadIdx.x], sh[threadIdx.x + o]); __syncthreads(); }
    mx = sh[0]; __syncthreads();
    float s = 0.0f;
    for (int j = threadIdx.x; j < L; j += 256) { float e = __expf(p[j * estride] - mx); p[j * estride] = e; s += e; }
    sh[threadIdx.x] = s; __syncthreads();
    for (int o = 128; o > 0; o >>= 1) { if (threadIdx.x < o) sh[threadIdx.x] += sh[threadIdx.x + o]; __syncthreads(); }
    float inv = 1.0f / sh[0]; __syncthreads();
    for (int j = threadIdx.x; j < L; j += 256) p[j * estride] *= inv;
}

// ======================================================================
// Elementwise / small reductions
// ======================================================================
__global__ void k_add(const float* a, const float* b, float* y, long long n)
{ long long i = blockIdx.x * 256ll + threadIdx.x; if (i < n) y[i] = a[i] + b[i]; }

__global__ void k_cvt16(const float* x, _Float16* y, long long n)
{ long long i = blockIdx.x * 256ll + threadIdx.x; if (i < n) y[i] = (_Float16)x[i]; }

__global__ void k_relu_tanh(const float* x, float* y, long long n)
{ long long i = blockIdx.x * 256ll + threadIdx.x; if (i < n) y[i] = tanhf(fmaxf(x[i], 0.0f)); }

__global__ void k_w8(const float* l, float* w, long long n)
{ long long i = blockIdx.x * 256ll + threadIdx.x; if (i < n) w[i] = fmaxf(tanhf(l[i]), 0.0f); }

__global__ void k_copyb(const float* src, float* dst, long long ss, long long ds, long long chunk, int batch)
{
    long long i = blockIdx.x * 256ll + threadIdx.x;
    long long n = chunk * batch;
    if (i >= n) return;
    int b = (int)(i / chunk);
    long long j = i % chunk;
    dst[b * ds + j] = src[b * ss + j];
}

__global__ __launch_bounds__(256) void k_abssum(const float* w, float* s, int N)
{
    __shared__ float sh[256];
    int b = blockIdx.x;
    float a = 0.0f;
    for (int n = threadIdx.x; n < N; n += 256) a += fabsf(w[(long long)b * N + n]);
    sh[threadIdx.x] = a; __syncthreads();
    for (int o = 128; o > 0; o >>= 1) { if (threadIdx.x < o) sh[threadIdx.x] += sh[threadIdx.x + o]; __syncthreads(); }
    if (threadIdx.x == 0) s[b] = sh[0];
}

__global__ void k_divmax(float* w, const float* s, int N, long long n)
{
    long long i = blockIdx.x * 256ll + threadIdx.x;
    if (i < n) { int b = (int)(i / N); w[i] /= fmaxf(s[b], 1e-12f); }
}

__global__ __launch_bounds__(256) void k_wsum(const float* x, const float* w, float* out, int C, int N)
{   // out[b*C+c] = sum_n x[b,c,n]*w[b,n]
    __shared__ float sh[256];
    int c = blockIdx.x, b = blockIdx.y;
    const float* xp = x + ((long long)b * C + c) * N;
    const float* wp = w + (long long)b * N;
    float a = 0.0f;
    for (int n = threadIdx.x; n < N; n += 256) a += xp[n] * wp[n];
    sh[threadIdx.x] = a; __syncthreads();
    for (int o = 128; o > 0; o >>= 1) { if (threadIdx.x < o) sh[threadIdx.x] += sh[threadIdx.x + o]; __syncthreads(); }
    if (threadIdx.x == 0) out[(long long)b * C + c] = sh[0];
}

__global__ void k_sigmul(const float* o, const float* x, float* y, int C, int L, long long n)
{   // y[b,c,l] = sigmoid(o[b,c]) * x[b,c,l]
    long long i = blockIdx.x * 256ll + threadIdx.x;
    if (i >= n) return;
    long long t = i / L;
    int c = (int)(t % C), b = (int)(t / C);
    float s = 1.0f / (1.0f + __expf(-o[(long long)b * C + c]));
    y[i] = s * x[i];
}

// ======================================================================
// weighted 8-point: XwX accumulation, 9x9 Jacobi eigh, episym residual
// ======================================================================
__global__ __launch_bounds__(256) void k_xwx(const float* xs, const float* w8, float* XwX, int N)
{
    __shared__ float sh[256];
    int pq = blockIdx.x, b = blockIdx.y;
    int p = pq / 9, q = pq % 9;
    float s = 0.0f;
    for (int n = threadIdx.x; n < N; n += 256) {
        const float* r = xs + ((long long)b * N + n) * 4;
        float x0 = r[0], x1 = r[1], x2 = r[2], x3 = r[3];
        float X[9] = { x2 * x0, x2 * x1, x2, x3 * x0, x3 * x1, x3, x0, x1, 1.0f };
        s += w8[(long long)b * N + n] * X[p] * X[q];
    }
    sh[threadIdx.x] = s; __syncthreads();
    for (int o = 128; o > 0; o >>= 1) { if (threadIdx.x < o) sh[threadIdx.x] += sh[threadIdx.x + o]; __syncthreads(); }
    if (threadIdx.x == 0) XwX[b * 81 + pq] = sh[0];
}

__global__ void k_eigh(const float* XwX, float* ehat, float* out_e, int Bn)
{
    int b = threadIdx.x;
    if (b >= Bn) return;
    float A[81], V[81];
    for (int i = 0; i < 81; ++i) { A[i] = XwX[b * 81 + i]; V[i] = 0.0f; }
    for (int i = 0; i < 9; ++i) V[i * 9 + i] = 1.0f;
    for (int sweep = 0; sweep < 40; ++sweep) {
        for (int p = 0; p < 8; ++p) for (int q = p + 1; q < 9; ++q) {
            float apq = A[p * 9 + q];
            if (fabsf(apq) < 1e-18f) continue;
            float app = A[p * 9 + p], aqq = A[q * 9 + q];
            float tau = (aqq - app) / (2.0f * apq);
            float t = (tau >= 0.0f ? 1.0f : -1.0f) / (fabsf(tau) + sqrtf(1.0f + tau * tau));
            float c = rsqrtf(1.0f + t * t), s = t * c;
            for (int i = 0; i < 9; ++i) {
                float aip = A[i * 9 + p], aiq = A[i * 9 + q];
                A[i * 9 + p] = c * aip - s * aiq;
                A[i * 9 + q] = s * aip + c * aiq;
            }
            for (int i = 0; i < 9; ++i) {
                float api = A[p * 9 + i], aqi = A[q * 9 + i];
                A[p * 9 + i] = c * api - s * aqi;
                A[q * 9 + i] = s * api + c * aqi;
            }
            for (int i = 0; i < 9; ++i) {
                float vip = V[i * 9 + p], viq = V[i * 9 + q];
                V[i * 9 + p] = c * vip - s * viq;
                V[i * 9 + q] = s * vip + c * viq;
            }
        }
    }
    int k = 0; float mv = A[0];
    for (int i = 1; i < 9; ++i) if (A[i * 9 + i] < mv) { mv = A[i * 9 + i]; k = i; }
    float nrm = 0.0f;
    for (int i = 0; i < 9; ++i) nrm += V[i * 9 + k] * V[i * 9 + k];
    nrm = rsqrtf(nrm + 1e-30f);
    for (int i = 0; i < 9; ++i) {
        float e = V[i * 9 + k] * nrm;
        ehat[b * 9 + i] = e;
        out_e[b * 9 + i] = e;
    }
}

__global__ void k_episym(const float* xs, const float* e, float* out, int N, int Bn)
{
    long long i = blockIdx.x * 256ll + threadIdx.x;
    if (i >= (long long)Bn * N) return;
    int b = (int)(i / N), n = (int)(i % N);
    const float* r = xs + ((long long)b * N + n) * 4;
    float x1 = r[0], y1 = r[1], x2 = r[2], y2 = r[3];
    const float* F = e + b * 9;
    float Fx0 = F[0] * x1 + F[1] * y1 + F[2];
    float Fx1 = F[3] * x1 + F[4] * y1 + F[5];
    float Fx2 = F[6] * x1 + F[7] * y1 + F[8];
    float Ft0 = F[0] * x2 + F[3] * y2 + F[6];
    float Ft1 = F[1] * x2 + F[4] * y2 + F[7];
    float d = x2 * Fx0 + y2 * Fx1 + Fx2;
    out[i] = d * d * (1.0f / (Fx0 * Fx0 + Fx1 * Fx1 + 1e-15f) +
                      1.0f / (Ft0 * Ft0 + Ft1 * Ft1 + 1e-15f));
}

// ======================================================================
// Host-side: parameter mapping (jax tree flatten = sorted dict keys),
// bump-allocator workspace, and graph orchestration.
// ======================================================================
namespace {

struct ConvP { const float* b; const float* w; };
struct BnP   { const float* b; const float* g; };
struct PcaP  { ConvP c1, c2; BnP c_bn; ConvP sp; BnP sp_bn; };
struct PcnP  { BnP bn1, bn2; ConvP c1, c2; PcaP ca; };
struct AttnP { ConvP cf1, cf2; BnP cf_bn; ConvP k, mh, q, v; };
struct OafP  { BnP bn1, bn2, bn3; ConvP c1, c2, c3; };
struct PoolP { BnP bn; ConvP c; };
struct Params {
    AttnP attn_a, attn_b;
    ConvP conv1;
    PoolP down;
    ConvP fuse; BnP fuse_bn;
    OafP oaf[6];
    ConvP out;
    PcnP pcn_a[3], pcn_b[3];
    PoolP up;
};

struct Cursor {
    void* const* din; int i;
    const float* f() { return (const float*)din[i++]; }
};
static ConvP rdConv(Cursor& c) { ConvP p; p.b = c.f(); p.w = c.f(); return p; }               // sorted: b, w
static BnP   rdBn(Cursor& c)   { BnP p; p.b = c.f(); p.g = c.f(); return p; }                 // sorted: b, g
static PcaP  rdPca(Cursor& c)  { PcaP p; p.c1 = rdConv(c); p.c2 = rdConv(c); p.c_bn = rdBn(c); p.sp = rdConv(c); p.sp_bn = rdBn(c); return p; }
static PcnP  rdPcn(Cursor& c)  { PcnP p; p.bn1 = rdBn(c); p.bn2 = rdBn(c); p.c1 = rdConv(c); p.c2 = rdConv(c); p.ca = rdPca(c); return p; }
static AttnP rdAttn(Cursor& c) { AttnP p; p.cf1 = rdConv(c); p.cf2 = rdConv(c); p.cf_bn = rdBn(c); p.k = rdConv(c); p.mh = rdConv(c); p.q = rdConv(c); p.v = rdConv(c); return p; }
static OafP  rdOaf(Cursor& c)  { OafP p; p.bn1 = rdBn(c); p.bn2 = rdBn(c); p.bn3 = rdBn(c); p.c1 = rdConv(c); p.c2 = rdConv(c); p.c3 = rdConv(c); return p; }
static PoolP rdPool(Cursor& c) { PoolP p; p.bn = rdBn(c); p.c = rdConv(c); return p; }

struct Ws {
    char* base; size_t off, cap;
    float* f(size_t n) { float* p = (float*)(base + off); off += ((n * sizeof(float) + 255) & ~(size_t)255); return p; }
    _Float16* h(size_t n) { _Float16* p = (_Float16*)(base + off); off += ((n * 2 + 255) & ~(size_t)255); return p; }
    size_t mark() const { return off; }
    void reset(size_t m) { off = m; }
};

static void gemm32(hipStream_t st, const float* A, long long abs_, int ars, int acs,
                   const float* Bm, long long bbs, int brs, int bcs,
                   float* D, long long dbs, int drs, int dcs,
                   const float* bias, int M, int N, int K, float scale, int relu, int batch)
{
    dim3 g((N + 15) / 16, (M + 127) / 128, batch);
    k_gemm_f32<<<g, 256, 0, st>>>(A, abs_, ars, acs, Bm, bbs, brs, bcs, D, dbs, drs, dcs, bias, M, N, K, scale, relu);
}

static void gemm16(hipStream_t st, const _Float16* A, long long abs_, int ars, int acs,
                   const _Float16* Bm, long long bbs, int brs, int bcs,
                   float* D, long long dbs, int drs, int dcs,
                   const float* bias, int M, int N, int K, float scale, int relu, int batch)
{
    dim3 g((N + 15) / 16, (M + 127) / 128, batch);
    k_gemm_f16<<<g, 256, 0, st>>>(A, abs_, ars, acs, Bm, bbs, brs, bcs, D, dbs, drs, dcs, bias, M, N, K, scale, relu);
}

// 1x1 conv == GEMM: y[b,o,l] = W[o,:] . x[b,:,l] + bias
static void convL(hipStream_t st, const ConvP& c, int Co, int Ci, const float* x, float* y, int L, int batch, int relu = 0)
{
    gemm32(st, c.w, 0, Ci, 1, x, (long long)Ci * L, L, 1, y, (long long)Co * L, L, 1, c.b, Co, L, Ci, 1.0f, relu, batch);
}

static void norm(hipStream_t st, Ws& ws, const float* x, float* y,
                 int CH, int L, long long sb, long long sc, long long sl,
                 int perBatch, float eps, const float* g, const float* bt, int relu, int batch)
{
    float* mean = ws.f(perBatch ? (size_t)batch * CH : (size_t)CH);
    float* var  = ws.f(perBatch ? (size_t)batch * CH : (size_t)CH);
    dim3 gs(CH, perBatch ? batch : 1);
    k_stats<<<gs, 256, 0, st>>>(x, sb, sc, sl, batch, L, perBatch, mean, var, CH);
    dim3 ga((L + 255) / 256, CH, batch);
    k_norm_apply<<<ga, 256, 0, st>>>(x, y, sb, sc, sl, CH, L, mean, var, perBatch, eps, g, bt, relu);
}

// ---------------- network blocks ----------------
static float* point_cn(hipStream_t st, Ws& ws, const PcnP& p, const float* x, int C, int N, int B)
{
    long long CN = (long long)C * N;
    float* out = ws.f((size_t)B * CN);
    size_t mk = ws.mark();
    float* o = ws.f((size_t)B * CN);
    convL(st, p.c1, C, C, x, o, N, B);
    norm(st, ws, o, o, C, N, CN, N, 1, 1, 1e-3f, nullptr, nullptr, 0, B);   // inorm
    norm(st, ws, o, o, C, N, CN, N, 1, 0, 1e-5f, p.bn1.g, p.bn1.b, 1, B);   // bn1 + relu
    {   // ---- point_ca ----
        const PcaP& ca = p.ca;
        int IC2 = C / 4;
        float* t = ws.f((size_t)B * CN);
        norm(st, ws, o, t, C, N, CN, N, 1, 1, 1e-3f, nullptr, nullptr, 0, B);
        norm(st, ws, t, t, C, N, CN, N, 1, 0, 1e-5f, ca.sp_bn.g, ca.sp_bn.b, 1, B);
        float* w = ws.f((size_t)B * N);
        convL(st, ca.sp, 1, C, t, w, N, B);
        k_relu_tanh<<<ceildiv_ll((long long)B * N, 256), 256, 0, st>>>(w, w, (long long)B * N);
        float* ssum = ws.f((size_t)B);
        k_abssum<<<B, 256, 0, st>>>(w, ssum, N);
        k_divmax<<<ceildiv_ll((long long)B * N, 256), 256, 0, st>>>(w, ssum, N, (long long)B * N);
        float* xsum = ws.f((size_t)B * C);
        { dim3 g(C, B); k_wsum<<<g, 256, 0, st>>>(o, w, xsum, C, N); }
        float* o2 = ws.f((size_t)B * IC2);
        gemm32(st, ca.c1.w, 0, C, 1, xsum, C, 1, 1, o2, IC2, 1, 1, ca.c1.b, IC2, 1, C, 1.0f, 0, B);
        norm(st, ws, o2, o2, IC2, 1, IC2, 1, 1, 0, 1e-5f, ca.c_bn.g, ca.c_bn.b, 1, B);
        float* o3 = ws.f((size_t)B * C);
        gemm32(st, ca.c2.w, 0, IC2, 1, o2, IC2, 1, 1, o3, C, 1, 1, ca.c2.b, C, 1, IC2, 1.0f, 0, B);
        k_sigmul<<<ceildiv_ll((long long)B * CN, 256), 256, 0, st>>>(o3, o, o, C, N, (long long)B * CN);
    }
    float* o4 = ws.f((size_t)B * CN);
    convL(st, p.c2, C, C, o, o4, N, B);
    norm(st, ws, o4, o4, C, N, CN, N, 1, 1, 1e-3f, nullptr, nullptr, 0, B);
    norm(st, ws, o4, o4, C, N, CN, N, 1, 0, 1e-5f, p.bn2.g, p.bn2.b, 1, B);
    k_add<<<ceildiv_ll((long long)B * CN, 256), 256, 0, st>>>(o4, x, out, (long long)B * CN);
    ws.reset(mk);
    return out;
}

static float* attn_prop(hipStream_t st, Ws& ws, const AttnP& ap, const float* x)
{
    const int C = 128, N = 2000, B = 2, H = 4, HD = 32;
    long long CN = (long long)C * N, NN = (long long)N * N;
    float* out = ws.f((size_t)B * CN);
    size_t mk = ws.mark();
    float* q = ws.f((size_t)B * CN);
    float* k = ws.f((size_t)B * CN);
    float* v = ws.f((size_t)B * CN);
    convL(st, ap.q, C, C, x, q, N, B);
    convL(st, ap.k, C, C, x, k, N, B);
    convL(st, ap.v, C, C, x, v, N, B);
    // pd = 2*descT.desc - xx_n - xx_m   (fp32 WMMA gram, scale=2)
    float* G = ws.f((size_t)B * NN);
    gemm32(st, x, CN, 1, N, x, CN, N, 1, G, NN, N, 1, nullptr, N, N, C, 2.0f, 0, B);
    float* xx = ws.f((size_t)B * N);
    { long long n = (long long)B * N; k_rowsumsq<<<ceildiv_ll(n, 256), 256, 0, st>>>(x, xx, C, N, n); }
    { long long n = (long long)B * NN; k_pd<<<ceildiv_ll(n, 256), 256, 0, st>>>(G, xx, N, n); }
    float* mask = ws.f((size_t)B * NN);
    { int rows = B * N; k_topk_mask<<<(rows + 7) / 8, 256, 0, st>>>(G, mask, N, 40, rows); }
    float* msym = ws.f((size_t)B * NN);
    { long long n = (long long)B * NN; k_masksym<<<ceildiv_ll(n, 256), 256, 0, st>>>(mask, msym, N, n); }
    // fp16 copies for the 16x16x32 WMMA path
    _Float16* q16 = ws.h((size_t)B * CN);
    _Float16* k16 = ws.h((size_t)B * CN);
    _Float16* v16 = ws.h((size_t)B * CN);
    k_cvt16<<<ceildiv_ll((long long)B * CN, 256), 256, 0, st>>>(q, q16, (long long)B * CN);
    k_cvt16<<<ceildiv_ll((long long)B * CN, 256), 256, 0, st>>>(k, k16, (long long)B * CN);
    k_cvt16<<<ceildiv_ll((long long)B * CN, 256), 256, 0, st>>>(v, v16, (long long)B * CN);
    float* addv = ws.f((size_t)B * CN);
    float* score = ws.f((size_t)NN);
    _Float16* score16 = ws.h((size_t)NN);
    const float sc = 0.17677669529663687f;  // 1/sqrt(32)
    for (int b = 0; b < B; ++b)
        for (int h = 0; h < H; ++h) {
            const _Float16* qh = q16 + (long long)b * CN + (long long)h * HD * N;
            const _Float16* kh = k16 + (long long)b * CN + (long long)h * HD * N;
            const _Float16* vh = v16 + (long long)b * CN + (long long)h * HD * N;
            // score[n,m] = qh[:,n].kh[:,m] * sc   (K=32 -> 1 WMMA per tile)
            gemm16(st, qh, 0, 1, N, kh, 0, N, 1, score, 0, N, 1, nullptr, N, N, HD, sc, 0, 1);
            k_mask_softmax<<<N, 256, 0, st>>>(score, msym + (long long)b * NN, N);
            k_cvt16<<<ceildiv_ll(NN, 256), 256, 0, st>>>(score, score16, NN);
            // addv[d,n] = sum_m v[d,m] * score[n,m]
            gemm16(st, vh, 0, N, 1, score16, 0, 1, N,
                   addv + (long long)b * CN + (long long)h * HD * N, 0, N, 1,
                   nullptr, HD, N, N, 1.0f, 0, 1);
        }
    float* addv2 = ws.f((size_t)B * CN);
    convL(st, ap.mh, C, C, addv, addv2, N, B);
    float* cat = ws.f((size_t)B * 2 * CN);
    k_copyb<<<ceildiv_ll((long long)B * CN, 256), 256, 0, st>>>(x, cat, CN, 2 * CN, CN, B);
    k_copyb<<<ceildiv_ll((long long)B * CN, 256), 256, 0, st>>>(addv2, cat + CN, CN, 2 * CN, CN, B);
    float* t1 = ws.f((size_t)B * 2 * CN);
    convL(st, ap.cf1, 2 * C, 2 * C, cat, t1, N, B);
    norm(st, ws, t1, t1, 2 * C, N, 2 * CN, N, 1, 0, 1e-5f, ap.cf_bn.g, ap.cf_bn.b, 1, B);
    float* t2 = ws.f((size_t)B * CN);
    convL(st, ap.cf2, C, 2 * C, t1, t2, N, B);
    k_add<<<ceildiv_ll((long long)B * CN, 256), 256, 0, st>>>(x, t2, out, (long long)B * CN);
    ws.reset(mk);
    return out;
}

static float* diff_pool(hipStream_t st, Ws& ws, const PoolP& p, const float* x, int C, int N, int P, int B)
{
    long long CN = (long long)C * N;
    float* out = ws.f((size_t)B * C * P);
    size_t mk = ws.mark();
    float* t = ws.f((size_t)B * CN);
    norm(st, ws, x, t, C, N, CN, N, 1, 1, 1e-3f, nullptr, nullptr, 0, B);
    norm(st, ws, t, t, C, N, CN, N, 1, 0, 1e-5f, p.bn.g, p.bn.b, 1, B);
    float* e = ws.f((size_t)B * P * N);
    convL(st, p.c, P, C, t, e, N, B);
    k_softmax_stride<<<B * P, 256, 0, st>>>(e, B * P, P, (long long)P * N, (long long)N, N, 1);
    // pooled[b,c,p] = sum_n x[b,c,n] * S[b,p,n]
    gemm32(st, x, CN, N, 1, e, (long long)P * N, 1, N, out, (long long)C * P, P, 1, nullptr, C, P, N, 1.0f, 0, B);
    ws.reset(mk);
    return out;
}

static float* diff_unpool(hipStream_t st, Ws& ws, const PoolP& p, const float* xup, const float* xdown,
                          int C, int N, int P, int B)
{
    long long CN = (long long)C * N;
    float* out = ws.f((size_t)B * CN);
    size_t mk = ws.mark();
    float* t = ws.f((size_t)B * CN);
    norm(st, ws, xup, t, C, N, CN, N, 1, 1, 1e-3f, nullptr, nullptr, 0, B);
    norm(st, ws, t, t, C, N, CN, N, 1, 0, 1e-5f, p.bn.g, p.bn.b, 1, B);
    float* e = ws.f((size_t)B * P * N);
    convL(st, p.c, P, C, t, e, N, B);
    // softmax over p (column stride N)
    k_softmax_stride<<<B * N, 256, 0, st>>>(e, B * N, N, (long long)P * N, 1, P, (long long)N);
    // out[b,c,n] = sum_p xdown[b,c,p] * S[b,p,n]
    gemm32(st, xdown, (long long)C * P, P, 1, e, (long long)P * N, N, 1, out, CN, N, 1, nullptr, C, N, P, 1.0f, 0, B);
    ws.reset(mk);
    return out;
}

static float* oafilter(hipStream_t st, Ws& ws, const OafP& p, const float* x, int C, int P, int B)
{
    long long CP = (long long)C * P;
    float* out = ws.f((size_t)B * CP);
    size_t mk = ws.mark();
    float* t = ws.f((size_t)B * CP);
    norm(st, ws, x, t, C, P, CP, P, 1, 1, 1e-3f, nullptr, nullptr, 0, B);
    norm(st, ws, t, t, C, P, CP, P, 1, 0, 1e-5f, p.bn1.g, p.bn1.b, 1, B);
    float* o = ws.f((size_t)B * CP);
    convL(st, p.c1, C, C, t, o, P, B);
    // transposed branch: bn over CH=P, L=C (strides swapped); conv over the P axis
    float* tr = ws.f((size_t)B * CP);
    norm(st, ws, o, tr, P, C, CP, 1, P, 0, 1e-5f, p.bn2.g, p.bn2.b, 1, B);
    float* o2 = ws.f((size_t)B * CP);
    // M=P,K=P,N=C; B[k,n]=tr[b,n,k]; D stored transposed back into (C,P) layout via d_rs=1,d_cs=P
    gemm32(st, p.c2.w, 0, P, 1, tr, CP, 1, P, o2, CP, 1, P, p.c2.b, P, C, P, 1.0f, 0, B);
    k_add<<<ceildiv_ll((long long)B * CP, 256), 256, 0, st>>>(o, o2, o, (long long)B * CP);
    norm(st, ws, o, t, C, P, CP, P, 1, 1, 1e-3f, nullptr, nullptr, 0, B);
    norm(st, ws, t, t, C, P, CP, P, 1, 0, 1e-5f, p.bn3.g, p.bn3.b, 1, B);
    float* o3 = ws.f((size_t)B * CP);
    convL(st, p.c3, C, C, t, o3, P, B);
    k_add<<<ceildiv_ll((long long)B * CP, 256), 256, 0, st>>>(o3, x, out, (long long)B * CP);
    ws.reset(mk);
    return out;
}

}  // namespace

extern "C" void kernel_launch(void* const* d_in, const int* in_sizes, int n_in,
                              void* d_out, int out_size, void* d_ws, size_t ws_size,
                              hipStream_t stream)
{
    (void)in_sizes; (void)n_in; (void)out_size;
    const int B = 2, C = 128, N = 2000, P = 500, IC = 4;
    const float* data = (const float*)d_in[0];
    const float* xs = (const float*)d_in[1];

    Cursor cur{ d_in, 2 };
    Params pr;
    pr.attn_a = rdAttn(cur);
    pr.attn_b = rdAttn(cur);
    pr.conv1 = rdConv(cur);
    pr.down = rdPool(cur);
    pr.fuse = rdConv(cur);
    pr.fuse_bn = rdBn(cur);
    for (int i = 0; i < 6; ++i) pr.oaf[i] = rdOaf(cur);
    pr.out = rdConv(cur);
    for (int i = 0; i < 3; ++i) pr.pcn_a[i] = rdPcn(cur);
    for (int i = 0; i < 3; ++i) pr.pcn_b[i] = rdPcn(cur);
    pr.up = rdPool(cur);

    Ws ws{ (char*)d_ws, 0, ws_size };
    long long CN = (long long)C * N;

    float* x = ws.f((size_t)B * CN);
    convL(stream, pr.conv1, C, IC, data, x, N, B);
    for (int i = 0; i < 3; ++i) x = point_cn(stream, ws, pr.pcn_a[i], x, C, N, B);
    float* x11 = attn_prop(stream, ws, pr.attn_a, x);
    float* x2 = diff_pool(stream, ws, pr.down, x11, C, N, P, B);
    for (int i = 0; i < 6; ++i) x2 = oafilter(stream, ws, pr.oaf[i], x2, C, P, B);
    float* xup = diff_unpool(stream, ws, pr.up, x11, x2, C, N, P, B);

    float* cat = ws.f((size_t)B * 2 * CN);
    k_copyb<<<ceildiv_ll((long long)B * CN, 256), 256, 0, stream>>>(x11, cat, CN, 2 * CN, CN, B);
    k_copyb<<<ceildiv_ll((long long)B * CN, 256), 256, 0, stream>>>(xup, cat + CN, CN, 2 * CN, CN, B);
    float* y = ws.f((size_t)B * CN);
    convL(stream, pr.fuse, C, 2 * C, cat, y, N, B);
    norm(stream, ws, y, y, C, N, CN, N, 1, 0, 1e-5f, pr.fuse_bn.g, pr.fuse_bn.b, 1, B);
    y = attn_prop(stream, ws, pr.attn_b, y);
    for (int i = 0; i < 3; ++i) y = point_cn(stream, ws, pr.pcn_b[i], y, C, N, B);

    // logits straight into d_out[0 .. B*N)
    float* logits = (float*)d_out;
    gemm32(stream, pr.out.w, 0, C, 1, y, CN, N, 1, logits, N, N, 1, pr.out.b, 1, N, C, 1.0f, 0, B);

    float* w8 = ws.f((size_t)B * N);
    k_w8<<<ceildiv_ll((long long)B * N, 256), 256, 0, stream>>>(logits, w8, (long long)B * N);
    float* XwX = ws.f((size_t)B * 81);
    { dim3 g(81, B); k_xwx<<<g, 256, 0, stream>>>(xs, w8, XwX, N); }
    float* ehat = ws.f((size_t)B * 9);
    k_eigh<<<1, 32, 0, stream>>>(XwX, ehat, (float*)d_out + (size_t)B * N, B);
    k_episym<<<ceildiv_ll((long long)B * N, 256), 256, 0, stream>>>(
        xs, ehat, (float*)d_out + (size_t)B * N + (size_t)B * 9, N, B);
}
